// Projector_63831803953166
// MI455X (gfx1250) — matrix operand order repaired and checked by reading, not compiled, not run on hardware
//
#include <hip/hip_runtime.h>
#include <hip/hip_bf16.h>

typedef __attribute__((ext_vector_type(16))) _Float16 v16h;
typedef __attribute__((ext_vector_type(8)))  float    v8f;
typedef __attribute__((ext_vector_type(2)))  __fp16   v2fp; // matches cvt_pkrtz return type

#define SIDE 128
#define TILES 8            // 128 / 16
#define CHUNK_K 32         // WMMA K for f16
#define ATOMS_PER_BLOCK 800

// Pack two f32 -> packed f16 dword with a single v_cvt_pk instruction.
__device__ __forceinline__ unsigned pack2(float a, float b) {
    v2fp p = __builtin_amdgcn_cvt_pkrtz(a, b);
    union { v2fp h; unsigned u; } r;
    r.h = p;
    return r.u;
}

__global__ void zero_f32_kernel(float* __restrict__ p, int n) {
    int i = blockIdx.x * blockDim.x + threadIdx.x;
    if (i < n) p[i] = 0.0f;
}

__global__ void __launch_bounds__(256)
proj_wmma_kernel(const float* __restrict__ mol, const float* __restrict__ stds,
                 const float* __restrict__ dens, float* __restrict__ out,
                 int A, int nsplit)
{
    // Atom parameters for the current K-chunk of 32 atoms
    // pNH holds -0.5/(var*ln2) so the Gaussian is exp2(pNH * d^2) -> bare v_exp_f32
    __shared__ __align__(16) float pX[CHUNK_K], pY[CHUNK_K], pNH[CHUNK_K], pCF[CHUNK_K];
    // f16 tiles in WMMA fragment order: [group 0..7][lane 0..31][vgpr 0..7] dwords
    __shared__ __align__(32) unsigned gxL[TILES * 32 * 8]; // B-matrix source (gx)
    __shared__ __align__(32) unsigned gyL[TILES * 32 * 8]; // A-matrix source (coef*gy)

    const int b  = blockIdx.x / nsplit;
    const int sp = blockIdx.x % nsplit;
    const int aBase = sp * ATOMS_PER_BLOCK;
    const int aEnd  = min(aBase + ATOMS_PER_BLOCK, A);
    const int nChunks = (aEnd - aBase + CHUNK_K - 1) / CHUNK_K;

    const int tid = threadIdx.x;
    const int l   = tid & 31;        // lane slot
    const int g   = tid >> 5;        // tile group (producer) == wave id (consumer)
    const int n   = l & 15;          // column/row within 16-wide tile
    // Pixel-center coordinate for this thread's column (gx) / row (gy):
    const float cw = (float)(16 * g + n) - 63.5f;

    v8f acc[TILES] = {};             // 8 output tiles per wave, f32

    const int kBaseB  = (l < 16) ? 0 : 16;  // B-matrix 32x16 f16 lane->K mapping
    const int kBaseA0 = (l < 16) ? 0 : 16;  // A-matrix 16x32 f16 lane->K mapping

    for (int c = 0; c < nChunks; ++c) {
        const int a0 = aBase + c * CHUNK_K;

        // ---- stage 0: stage atom params in LDS (32 threads) ----
        if (tid < CHUNK_K) {
            int a = a0 + tid;
            bool ok = a < aEnd;
            int ai = ok ? a : 0;
            size_t base = ((size_t)b * (size_t)A + (size_t)ai) * 3;
            float x = mol[base + 0];
            float y = mol[base + 1];
            float s = stds[ai];
            float rv = __builtin_amdgcn_rcpf(s * s);   // 1/var, single v_rcp_f32
            pX[tid]  = x;
            pY[tid]  = y;
            // -0.5/(var*ln2): exp(-d^2/(2 var)) == exp2(pNH * d^2)
            pNH[tid] = -0.72134752044448170368f * rv;
            // coef = dens/(2*pi*var); zeroed for tail atoms so they contribute 0
            pCF[tid] = ok ? dens[ai] * 0.15915494309189535f * rv : 0.0f;
        }
        __syncthreads();

        // ---- stage 1: produce f16 fragments in LDS (each exp computed once) ----
        unsigned* dx = &gxL[(g * 32 + l) * 8];
        unsigned* dy = &gyL[(g * 32 + l) * 8];
#pragma unroll
        for (int j = 0; j < 8; ++j) {
            // gx, B-matrix layout: lane l holds N = l%16, K = kBaseB + 2j, 2j+1
            int k0 = kBaseB + 2 * j;
            float d0 = cw - pX[k0];
            float d1 = cw - pX[k0 + 1];
            float e0 = __builtin_amdgcn_exp2f(pNH[k0]     * d0 * d0);
            float e1 = __builtin_amdgcn_exp2f(pNH[k0 + 1] * d1 * d1);
            dx[j] = pack2(e0, e1);
        }
#pragma unroll
        for (int j = 0; j < 8; ++j) {
            // coef*gy, A-matrix layout: lane l holds M = l%16,
            // K = kBaseA0 + 2j, 2j+1 (same contiguous order as B)
            int k0 = kBaseA0 + 2 * j;
            float d0 = cw - pY[k0];
            float d1 = cw - pY[k0 + 1];
            float e0 = pCF[k0]     * __builtin_amdgcn_exp2f(pNH[k0]     * d0 * d0);
            float e1 = pCF[k0 + 1] * __builtin_amdgcn_exp2f(pNH[k0 + 1] * d1 * d1);
            dy[j] = pack2(e0, e1);
        }
        __syncthreads();

        // ---- stage 2: WMMA — wave g owns h-tile g, sweeps 8 w-tiles ----
        const v16h af = *reinterpret_cast<const v16h*>(&gyL[(g * 32 + l) * 8]);
#pragma unroll
        for (int t = 0; t < TILES; ++t) {
            const v16h bf = *reinterpret_cast<const v16h*>(&gxL[(t * 32 + l) * 8]);
            acc[t] = __builtin_amdgcn_wmma_f32_16x16x32_f16(
                false, af, false, bf, (short)0, acc[t], false, false);
        }
        __syncthreads();  // protect LDS before next chunk's producers
    }

    // ---- epilogue: f32 atomic accumulate across atom splits ----
    // D layout: lanes 0-15 -> M = v, lanes 16-31 -> M = 8+v; N = l%16
    const int mBase = (l < 16) ? 0 : 8;
    float* outB = out + (size_t)b * SIDE * SIDE;
#pragma unroll
    for (int t = 0; t < TILES; ++t) {
#pragma unroll
        for (int v = 0; v < 8; ++v) {
            int h = 16 * g + mBase + v;
            int w = 16 * t + n;
            atomicAdd(&outB[h * SIDE + w], acc[t][v]);
        }
    }
}

extern "C" void kernel_launch(void* const* d_in, const int* in_sizes, int n_in,
                              void* d_out, int out_size, void* d_ws, size_t ws_size,
                              hipStream_t stream) {
    const float* mol  = (const float*)d_in[0];  // (B, A, 3) f32
    const float* stds = (const float*)d_in[1];  // (A,)     f32
    const float* dens = (const float*)d_in[2];  // (A,)     f32
    float* out = (float*)d_out;                 // (B, 128, 128) f32

    const int A = in_sizes[1];
    const int B = (A > 0) ? (in_sizes[0] / (3 * A)) : 0;

    // Zero output (harness poisons it; we accumulate with atomics)
    zero_f32_kernel<<<(out_size + 255) / 256, 256, 0, stream>>>(out, out_size);

    const int nsplit = (A + ATOMS_PER_BLOCK - 1) / ATOMS_PER_BLOCK;  // 25 for A=20000
    dim3 grid(B * nsplit);                                           // 400 blocks
    proj_wmma_kernel<<<grid, 256, 0, stream>>>(mol, stds, dens, out, A, nsplit);
}
